// RadarSparseBackbone_35716948033611
// MI455X (gfx1250) — matrix operand order repaired and checked by reading, not compiled, not run on hardware
//
#include <hip/hip_runtime.h>
#include <hip/hip_bf16.h>
#include <stdint.h>

// ---------------- problem constants (from reference) ----------------
#define Z0   10
#define Y0   200
#define X0   704
#define Z1D  5
#define Y1D  100
#define X1D  352
#define NIN  20000                 // B_AG * N_PTS
#define CAP0 540000                // 27 * NIN
#define CAP1 352000                // B_AG * Z1*Y1*X1
#define HW   (Y0 * X0)             // 140800
#define EPSF 1e-5f
#define NB   128                   // pair-blocks per offset k in spconv grid

typedef __attribute__((ext_vector_type(16))) __bf16 v16bf;
typedef __attribute__((ext_vector_type(8)))  float  v8f;

__device__ __forceinline__ unsigned short f2bf(float f) {
  unsigned int u = __float_as_uint(f);
  unsigned int r = u + 0x7FFFu + ((u >> 16) & 1u);   // round-to-nearest-even
  return (unsigned short)(r >> 16);
}

union AFrag {
  v16bf v;
  unsigned short u[16];
  uint4 q[2];
};

// ---------------- input GEMM: f = sp_feats[20000x64] @ w_in[64x32] -> bf16 ----
__global__ __launch_bounds__(256)
void in_gemm_kernel(const float* __restrict__ feats, const float* __restrict__ Wm,
                    unsigned short* __restrict__ outh)
{
  const int wave = threadIdx.x >> 5, lane = threadIdx.x & 31;
  const int n = lane & 15, hi = lane >> 4;

  AFrag B[2][2];
  #pragma unroll
  for (int nt = 0; nt < 2; ++nt)
    #pragma unroll
    for (int kk = 0; kk < 2; ++kk)
      #pragma unroll
      for (int e = 0; e < 16; ++e)
        B[nt][kk].u[e] = f2bf(Wm[(size_t)(kk * 32 + hi * 8 + (e & 7) + ((e >> 3) << 4)) * 32 + nt * 16 + n]);

  const int nTiles = NIN >> 4;
  for (int t = blockIdx.x * 8 + wave; t < nTiles; t += (int)gridDim.x * 8) {
    const int p0 = t << 4;
    const int m  = lane & 15;
    const float* row = feats + (size_t)(p0 + m) * 64;
    AFrag A[2];
    #pragma unroll
    for (int kk = 0; kk < 2; ++kk) {
      const int base = kk * 32 + hi * 8;
      #pragma unroll
      for (int j = 0; j < 8; ++j) {
        A[kk].u[j]     = f2bf(row[base + j]);
        A[kk].u[8 + j] = f2bf(row[base + 16 + j]);
      }
    }
    #pragma unroll
    for (int nt = 0; nt < 2; ++nt) {
      v8f c = {};
      #pragma unroll
      for (int kk = 0; kk < 2; ++kk)
        c = __builtin_amdgcn_wmma_f32_16x16x32_bf16(false, A[kk].v, false, B[nt][kk].v,
                                                    (short)0, c, false, false);
      #pragma unroll
      for (int r = 0; r < 8; ++r)
        outh[(size_t)(p0 + r + hi * 8) * 32 + nt * 16 + n] = f2bf(c[r]);
    }
  }
}

// ---------------- sparse conv: gather(16 pairs) -> WMMA -> atomic scatter ------
template<int CIN, int COUT>
__global__ __launch_bounds__(256)
void spconv_wmma(const unsigned short* __restrict__ feats,
                 const int* __restrict__ g, const int* __restrict__ s,
                 const float* __restrict__ W, int P, int outCap,
                 float* __restrict__ acc)
{
  constexpr int NKT = CIN / 32;
  constexpr int NNT = COUT / 16;
  const int k    = blockIdx.x / NB;      // 0..26 kernel offset
  const int blk  = blockIdx.x % NB;
  const int wave = threadIdx.x >> 5;
  const int lane = threadIdx.x & 31;
  const int n    = lane & 15;
  const int hi   = lane >> 4;
  const float* Wk = W + (size_t)k * CIN * COUT;

  // B fragments (per-offset weight) resident for the whole wave
  AFrag B[NNT][NKT];
  #pragma unroll
  for (int nt = 0; nt < NNT; ++nt)
    #pragma unroll
    for (int kk = 0; kk < NKT; ++kk)
      #pragma unroll
      for (int e = 0; e < 16; ++e)
        B[nt][kk].u[e] = f2bf(Wk[(size_t)(kk * 32 + hi * 8 + (e & 7) + ((e >> 3) << 4)) * COUT + nt * 16 + n]);

  const int* gk = g + (size_t)k * P;
  const int* sk = s + (size_t)k * P;
  const int nTiles = P >> 4;             // P is a multiple of 16 for all layers

  for (int t = blk * 8 + wave; t < nTiles; t += NB * 8) {
    const int p0 = t << 4;
    const int m  = lane & 15;
    const int gi = gk[p0 + m];           // dummy pairs gather row 0 (harmless)
    const uint4* row = reinterpret_cast<const uint4*>(feats + (size_t)gi * CIN);
    AFrag A[NKT];
    #pragma unroll
    for (int kk = 0; kk < NKT; ++kk) {
      const int qb = kk * 4 + hi;        // (kk*32 + hi*8)/8 in uint4 units
      A[kk].q[0] = row[qb];
      A[kk].q[1] = row[qb + 2];          // +16 bf16 elements
    }
    #pragma unroll
    for (int nt = 0; nt < NNT; ++nt) {
      v8f c = {};
      #pragma unroll
      for (int kk = 0; kk < NKT; ++kk)
        c = __builtin_amdgcn_wmma_f32_16x16x32_bf16(false, A[kk].v, false, B[nt][kk].v,
                                                    (short)0, c, false, false);
      #pragma unroll
      for (int r = 0; r < 8; ++r) {
        const int si = sk[p0 + r + hi * 8];
        if (si < outCap)                 // s==outCap -> reference's dropped trash row
          atomicAdd(acc + (size_t)si * COUT + nt * 16 + n, c[r]);
      }
    }
  }
}

// ---------------- BN + ReLU, emit bf16 (next spconv) and/or f32 (ConvT) -------
__global__ void bn_relu_kernel(const float* __restrict__ acc, const float* __restrict__ bnp,
                               long long total, int C,
                               unsigned short* __restrict__ oh, float* __restrict__ of)
{
  long long i      = (long long)blockIdx.x * blockDim.x + threadIdx.x;
  long long stride = (long long)gridDim.x * blockDim.x;
  for (; i < total; i += stride) {
    int c = (int)(i % C);
    float gg = bnp[c], bb = bnp[C + c], mm = bnp[2 * C + c], vv = bnp[3 * C + c];
    float y = (acc[i] - mm) * (gg * rsqrtf(vv + EPSF)) + bb;
    y = fmaxf(y, 0.f);
    if (oh) oh[i] = f2bf(y);
    if (of) of[i] = y;
  }
}

// ---------------- fused to_dense + ConvTranspose2d k=1 s=1 (stage 0) ----------
__global__ void convt0_kernel(const float* __restrict__ F, const int* __restrict__ coords,
                              const float* __restrict__ Wct, float* __restrict__ bev)
{
  long long total  = (long long)CAP0 * 64;
  long long id     = (long long)blockIdx.x * blockDim.x + threadIdx.x;
  long long stride = (long long)gridDim.x * blockDim.x;
  for (; id < total; id += stride) {
    const int row = (int)(id >> 6);
    const int o   = (int)(id & 63);
    const int* cd = coords + (size_t)row * 4;
    const int b = cd[0], z = cd[1], y = cd[2], x = cd[3];
    if (z >= Z0) continue;               // pad coords dropped (out-of-bounds scatter)
    const float* fr = F + (size_t)row * 32;
    float sum = 0.f;
    #pragma unroll
    for (int ci = 0; ci < 32; ++ci)
      sum += fr[ci] * Wct[(size_t)(ci * Z0 + z) * 64 + o];
    atomicAdd(bev + ((size_t)(b * 64 + o) * Y0 + y) * X0 + x, sum);
  }
}

// ---------------- fused to_dense + ConvTranspose2d k=2 s=2 (stage 1) ----------
__global__ void convt1_kernel(const float* __restrict__ F, const int* __restrict__ coords,
                              const float* __restrict__ Wct, float* __restrict__ bev)
{
  long long total  = (long long)CAP1 * 64;
  long long id     = (long long)blockIdx.x * blockDim.x + threadIdx.x;
  long long stride = (long long)gridDim.x * blockDim.x;
  for (; id < total; id += stride) {
    const int row = (int)(id >> 6);
    const int o   = (int)(id & 63);
    const int* cd = coords + (size_t)row * 4;
    const int b = cd[0], z = cd[1], y = cd[2], x = cd[3];
    if (z >= Z1D) continue;
    const float* fr = F + (size_t)row * 64;
    float a00 = 0.f, a01 = 0.f, a10 = 0.f, a11 = 0.f;
    #pragma unroll 8
    for (int ci = 0; ci < 64; ++ci) {
      const float fv = fr[ci];
      const float* wp = Wct + ((size_t)(ci * Z1D + z) * 64 + o) * 4; // [.., o, ky, kx]
      a00 += fv * wp[0]; a01 += fv * wp[1];
      a10 += fv * wp[2]; a11 += fv * wp[3];
    }
    const int oy = 2 * y, ox = 2 * x;
    float* base = bev + (size_t)(b * 64 + o) * Y0 * X0;
    atomicAdd(base + (size_t)oy * X0 + ox,           a00);
    atomicAdd(base + (size_t)oy * X0 + ox + 1,       a01);
    atomicAdd(base + (size_t)(oy + 1) * X0 + ox,     a10);
    atomicAdd(base + (size_t)(oy + 1) * X0 + ox + 1, a11);
  }
}

// ---------------- bias + BN + ReLU + 2-agent attention fusion (wave/pixel) ----
__global__ __launch_bounds__(256)
void fusion_kernel(const float* __restrict__ pre, const float* __restrict__ bias,
                   const float* __restrict__ bnp, float* __restrict__ out)
{
  const int wave = threadIdx.x >> 5, lane = threadIdx.x & 31;
  const int pix  = blockIdx.x * 8 + wave;
  if (pix >= HW) return;
  const int c0 = lane, c1 = lane + 32;

  auto bnrelu = [&](float v, int c) {
    float gg = bnp[c], bb = bnp[64 + c], mm = bnp[128 + c], vv = bnp[192 + c];
    return fmaxf((v - mm) * (gg * rsqrtf(vv + EPSF)) + bb, 0.f);
  };
  float x0a = bnrelu(pre[(size_t)c0 * HW + pix]        + bias[c0], c0);
  float x0b = bnrelu(pre[(size_t)c1 * HW + pix]        + bias[c1], c1);
  float x1a = bnrelu(pre[(size_t)(64 + c0) * HW + pix] + bias[c0], c0);
  float x1b = bnrelu(pre[(size_t)(64 + c1) * HW + pix] + bias[c1], c1);

  float d00 = x0a * x0a + x0b * x0b;     // x0.x0 partial
  float d01 = x0a * x1a + x0b * x1b;     // x0.x1 partial
  #pragma unroll
  for (int mlen = 16; mlen >= 1; mlen >>= 1) {
    d00 += __shfl_xor(d00, mlen, 32);
    d01 += __shfl_xor(d01, mlen, 32);
  }
  const float s00 = d00 * 0.125f, s01 = d01 * 0.125f;  // / sqrt(64)
  const float mx = fmaxf(s00, s01);
  const float e0 = __expf(s00 - mx), e1 = __expf(s01 - mx);
  const float inv = 1.f / (e0 + e1);
  const float a0 = e0 * inv, a1 = e1 * inv;
  out[(size_t)c0 * HW + pix] = a0 * x0a + a1 * x1a;    // keep ego (agent 0)
  out[(size_t)c1 * HW + pix] = a0 * x0b + a1 * x1b;
}

// ------------------------------- launch ---------------------------------------
extern "C" void kernel_launch(void* const* d_in, const int* in_sizes, int n_in,
                              void* d_out, int out_size, void* d_ws, size_t ws_size,
                              hipStream_t stream)
{
  (void)in_sizes; (void)n_in; (void)out_size; (void)ws_size;
  const float* sp_feats = (const float*)d_in[0];
  const float* w_in  = (const float*)d_in[1];
  const float* w0    = (const float*)d_in[2];
  const float* bnp0  = (const float*)d_in[3];
  const float* w0a   = (const float*)d_in[4];
  const float* bnp0a = (const float*)d_in[5];
  const float* w0b   = (const float*)d_in[6];
  const float* bnp0b = (const float*)d_in[7];
  const float* w1    = (const float*)d_in[8];
  const float* bnp1  = (const float*)d_in[9];
  const float* w1a   = (const float*)d_in[10];
  const float* bnp1a = (const float*)d_in[11];
  const float* w1b   = (const float*)d_in[12];
  const float* bnp1b = (const float*)d_in[13];
  const float* ct0_w = (const float*)d_in[14];
  const float* ct0_b = (const float*)d_in[15];
  const float* bnt0  = (const float*)d_in[16];
  const float* ct1_w = (const float*)d_in[17];
  const float* ct1_b = (const float*)d_in[18];
  const float* bnt1  = (const float*)d_in[19];
  const int* coords0 = (const int*)d_in[20];
  const int* coords1 = (const int*)d_in[21];
  const int* g0 = (const int*)d_in[22];
  const int* s0 = (const int*)d_in[23];
  const int* ga = (const int*)d_in[24];
  const int* sa = (const int*)d_in[25];
  const int* g1 = (const int*)d_in[26];
  const int* s1 = (const int*)d_in[27];
  const int* gb = (const int*)d_in[28];
  const int* sb = (const int*)d_in[29];

  // workspace layout (all offsets 256B aligned)
  constexpr size_t SZ_A0  = (size_t)CAP0 * 32 * 4;   // 69,120,000
  constexpr size_t SZ_A2  = (size_t)CAP1 * 64 * 4;   // 90,112,000
  constexpr size_t SZ_B0H = (size_t)CAP0 * 32 * 2;   // 34,560,000
  constexpr size_t SZ_B2H = (size_t)CAP1 * 64 * 2;   // 45,056,000
  constexpr size_t SZ_BEV = (size_t)2 * 64 * HW * 4; // 72,089,600
  char* ws = (char*)d_ws;
  size_t off = 0;
  float* A0 = (float*)(ws + off); off += SZ_A0;
  float* A1 = (float*)(ws + off); off += SZ_A0;
  float* F0 = (float*)(ws + off); off += SZ_A0;
  float* A2 = (float*)(ws + off); off += SZ_A2;
  float* A3 = (float*)(ws + off); off += SZ_A2;
  float* F1 = (float*)(ws + off); off += SZ_A2;
  unsigned short* B0h = (unsigned short*)(ws + off); off += SZ_B0H;
  unsigned short* B1h = (unsigned short*)(ws + off); off += SZ_B0H;
  unsigned short* B2h = (unsigned short*)(ws + off); off += SZ_B2H;
  unsigned short* B3h = (unsigned short*)(ws + off); off += SZ_B2H;
  float* bev0 = (float*)(ws + off); off += SZ_BEV;
  float* bev1 = (float*)(ws + off); off += SZ_BEV;
  unsigned short* Fh = (unsigned short*)(ws + off); off += (size_t)NIN * 32 * 2;

  float* out = (float*)d_out;
  const dim3 blk(256);
  const int bnBlk0 = (int)(((long long)CAP0 * 32 + 255) / 256);
  const int bnBlk1 = (int)(((long long)CAP1 * 64 + 255) / 256);

  // input GEMM (f32 -> bf16 activations)
  in_gemm_kernel<<<160, blk, 0, stream>>>(sp_feats, w_in, Fh);

  // stage 0: spconv + 2x subm, all C=32
  hipMemsetAsync(A0, 0, SZ_A0, stream);
  spconv_wmma<32, 32><<<27 * NB, blk, 0, stream>>>(Fh, g0, s0, w0, NIN, CAP0, A0);
  bn_relu_kernel<<<bnBlk0, blk, 0, stream>>>(A0, bnp0, (long long)CAP0 * 32, 32, B0h, nullptr);

  hipMemsetAsync(A1, 0, SZ_A0, stream);
  spconv_wmma<32, 32><<<27 * NB, blk, 0, stream>>>(B0h, ga, sa, w0a, CAP0, CAP0, A1);
  bn_relu_kernel<<<bnBlk0, blk, 0, stream>>>(A1, bnp0a, (long long)CAP0 * 32, 32, B1h, nullptr);

  hipMemsetAsync(A0, 0, SZ_A0, stream);
  spconv_wmma<32, 32><<<27 * NB, blk, 0, stream>>>(B1h, ga, sa, w0b, CAP0, CAP0, A0);
  bn_relu_kernel<<<bnBlk0, blk, 0, stream>>>(A0, bnp0b, (long long)CAP0 * 32, 32, B0h, F0);

  // BEV stage 0: scatter ConvT(1x1) + fusion -> out channels [0,64)
  hipMemsetAsync(bev0, 0, SZ_BEV, stream);
  convt0_kernel<<<135000, blk, 0, stream>>>(F0, coords0, ct0_w, bev0);
  fusion_kernel<<<(HW + 7) / 8, blk, 0, stream>>>(bev0, ct0_b, bnt0, out);

  // stage 1: strided spconv 32->64 + 2x subm 64->64
  hipMemsetAsync(A2, 0, SZ_A2, stream);
  spconv_wmma<32, 64><<<27 * NB, blk, 0, stream>>>(B0h, g1, s1, w1, CAP0, CAP1, A2);
  bn_relu_kernel<<<bnBlk1, blk, 0, stream>>>(A2, bnp1, (long long)CAP1 * 64, 64, B2h, nullptr);

  hipMemsetAsync(A3, 0, SZ_A2, stream);
  spconv_wmma<64, 64><<<27 * NB, blk, 0, stream>>>(B2h, gb, sb, w1a, CAP1, CAP1, A3);
  bn_relu_kernel<<<bnBlk1, blk, 0, stream>>>(A3, bnp1a, (long long)CAP1 * 64, 64, B3h, nullptr);

  hipMemsetAsync(A2, 0, SZ_A2, stream);
  spconv_wmma<64, 64><<<27 * NB, blk, 0, stream>>>(B3h, gb, sb, w1b, CAP1, CAP1, A2);
  bn_relu_kernel<<<bnBlk1, blk, 0, stream>>>(A2, bnp1b, (long long)CAP1 * 64, 64, nullptr, F1);

  // BEV stage 1: scatter ConvT(2x2,s=2) + fusion -> out channels [64,128)
  hipMemsetAsync(bev1, 0, SZ_BEV, stream);
  convt1_kernel<<<88000, blk, 0, stream>>>(F1, coords1, ct1_w, bev1);
  fusion_kernel<<<(HW + 7) / 8, blk, 0, stream>>>(bev1, ct1_b, bnt1, out + (size_t)64 * HW);
}
